// RNNManual_30812095381739
// MI455X (gfx1250) — matrix operand order repaired and checked by reading, not compiled, not run on hardware
//
#include <hip/hip_runtime.h>
#include <hip/hip_bf16.h>
#include <math.h>

// ---------------------------------------------------------------------------
// RNN forward on MI455X (gfx1250), wave32 + WMMA bf16 (f32 accumulate).
//   pre = x @ (Wi@We)^T + bih          (one big 65536x512x512 GEMM, WMMA)
//   scan: h = mask(tanh(pre_t + h@Wh^T))   (8 persistent WGs, h in LDS)
//   out = hT @ Who^T + bho             (epilogue of scan kernel)
// h kept twice in LDS: fp32 (output head) + bf16 mirror (pure-DS A fragments).
// ---------------------------------------------------------------------------

typedef __attribute__((ext_vector_type(16))) __bf16          v16bf;
typedef __attribute__((ext_vector_type(8)))  float           v8f;
typedef __attribute__((ext_vector_type(8)))  unsigned short  u16x8;
typedef __attribute__((ext_vector_type(16))) unsigned short  u16x16;

#define RNN_B      128
#define RNN_T      512
#define RNN_IN     512
#define RNN_EMB    256
#define RNN_H      512
#define RNN_OUT    8
#define HSTRIDE    516   // fp32 h row stride (floats)  -> bank shift 4/row
#define HBSTRIDE   520   // bf16 h row stride (elems)   -> bank shift 4/row, 16B aligned

// fp32 -> bf16 via native convert
static __device__ __forceinline__ unsigned short f2bf(float f) {
    __bf16 b = (__bf16)f;
    return __builtin_bit_cast(unsigned short, b);
}

static __device__ __forceinline__ v16bf pack16(u16x8 lo, u16x8 hi) {
    u16x16 r = __builtin_shufflevector(lo, hi, 0,1,2,3,4,5,6,7,8,9,10,11,12,13,14,15);
    return __builtin_bit_cast(v16bf, r);
}

// A fragment (16x32 bf16): lane<16 -> row=lane, k={0..7,16..23};
// lane>=16 -> row=lane-16, k={8..15,24..31}.  Row-major src, stride 512.
static __device__ __forceinline__ v16bf load_fragA_bf16(const unsigned short* p,
                                                        int m0, int k0, int lane) {
    int row = m0 + (lane & 15);
    int kh  = k0 + ((lane >> 4) << 3);          // +0 or +8
    const unsigned short* base = p + row * 512 + kh;
    u16x8 c0 = *(const u16x8*)(base);
    u16x8 c1 = *(const u16x8*)(base + 16);
    return pack16(c0, c1);
}

// A fragment from bf16 h mirror in LDS (row stride HBSTRIDE).
static __device__ __forceinline__ v16bf load_fragA_lds(const unsigned short* hb,
                                                       int k0, int lane) {
    int row = lane & 15;
    int kh  = k0 + ((lane >> 4) << 3);
    const unsigned short* base = hb + row * HBSTRIDE + kh;
    u16x8 c0 = *(const u16x8*)(base);
    u16x8 c1 = *(const u16x8*)(base + 16);
    return pack16(c0, c1);
}

// B fragment (32x16 bf16, KxN): lane = column n; lanes 0-15 k={0..7,16..23},
// lanes 16-31 k={8..15,24..31}.  Src is W[n,k] row-major (= B^T), stride 512.
static __device__ __forceinline__ v16bf load_fragB_bf16(const unsigned short* p,
                                                        int n0, int k0, int lane) {
    int col = n0 + (lane & 15);
    int kh  = k0 + ((lane >> 4) << 3);          // +0 or +8
    const unsigned short* base = p + col * 512 + kh;
    u16x8 c0 = *(const u16x8*)(base);
    u16x8 c1 = *(const u16x8*)(base + 16);
    return pack16(c0, c1);
}

static __device__ __forceinline__ v8f wmma_bf16(v16bf a, v16bf b, v8f c) {
    return __builtin_amdgcn_wmma_f32_16x16x32_bf16(false, a, false, b,
                                                   (short)0, c, false, false);
}

// ---------------------------------------------------------------------------
// Kernel 1: W = Wi @ We (fused projection weight) -> bf16; Wh -> bf16.
// ---------------------------------------------------------------------------
__global__ void fuse_weights_kernel(const float* __restrict__ Wih,
                                    const float* __restrict__ We,
                                    unsigned short* __restrict__ Wbf,
                                    unsigned short* __restrict__ Whbf) {
    int idx = blockIdx.x * 256 + threadIdx.x;        // 512*512 threads
    if (idx >= RNN_H * RNN_IN) return;
    int n = idx >> 9;
    int v = idx & 511;
    float s = 0.f;
    const float* wr = Wih + n * (RNN_EMB + RNN_H);
    #pragma unroll 4
    for (int e = 0; e < RNN_EMB; ++e) s += wr[e] * We[e * RNN_IN + v];
    Wbf[idx]  = f2bf(s);
    Whbf[idx] = f2bf(wr[RNN_EMB + v]);               // Wh[n, v]
}

// ---------------------------------------------------------------------------
// Kernel 2: x fp32 -> bf16, 8 elements / thread (b128 in, b128 out)
// ---------------------------------------------------------------------------
__global__ void cvt_x_kernel(const float* __restrict__ x,
                             unsigned short* __restrict__ xbf, int n8) {
    int i = blockIdx.x * 256 + threadIdx.x;
    if (i >= n8) return;
    const float4* p = (const float4*)(x + (size_t)i * 8);
    float4 a = p[0], b = p[1];
    u16x8 r;
    r[0] = f2bf(a.x); r[1] = f2bf(a.y); r[2] = f2bf(a.z); r[3] = f2bf(a.w);
    r[4] = f2bf(b.x); r[5] = f2bf(b.y); r[6] = f2bf(b.z); r[7] = f2bf(b.w);
    *(u16x8*)(xbf + (size_t)i * 8) = r;
}

// ---------------------------------------------------------------------------
// Kernel 3: pre = x @ W^T + bih, stored transposed as pre[t][b][n].
// Block tile 128(M) x 64(N), 8 waves, each wave 32x32 (2x2 WMMA tiles).
// ---------------------------------------------------------------------------
__global__ __launch_bounds__(256)
void gemm_pre_kernel(const unsigned short* __restrict__ xbf,
                     const unsigned short* __restrict__ Wbf,
                     const float* __restrict__ bih,
                     float* __restrict__ pre) {
    int lane = threadIdx.x & 31;
    int w    = threadIdx.x >> 5;
    int m0   = blockIdx.y * 128 + (w & 3) * 32;      // 4 waves along M
    int n0   = blockIdx.x * 64  + (w >> 2) * 32;     // 2 waves along N

    v8f acc00 = {}, acc01 = {}, acc10 = {}, acc11 = {};
    #pragma unroll 4
    for (int k0 = 0; k0 < RNN_IN; k0 += 32) {
        v16bf a0 = load_fragA_bf16(xbf, m0,      k0, lane);
        v16bf a1 = load_fragA_bf16(xbf, m0 + 16, k0, lane);
        v16bf b0 = load_fragB_bf16(Wbf, n0,      k0, lane);
        v16bf b1 = load_fragB_bf16(Wbf, n0 + 16, k0, lane);
        acc00 = wmma_bf16(a0, b0, acc00);
        acc01 = wmma_bf16(a0, b1, acc01);
        acc10 = wmma_bf16(a1, b0, acc10);
        acc11 = wmma_bf16(a1, b1, acc11);
    }

    // Epilogue: row = b*T + t  ->  store at pre[(t*B + b)*H + n], add bias.
    v8f* accs[2][2] = { {&acc00, &acc01}, {&acc10, &acc11} };
    #pragma unroll
    for (int mt = 0; mt < 2; ++mt) {
        int rbase = m0 + mt * 16 + ((lane >> 4) << 3);
        #pragma unroll
        for (int nt = 0; nt < 2; ++nt) {
            int col  = n0 + nt * 16 + (lane & 15);
            float bv = bih[col];
            v8f a = *accs[mt][nt];
            #pragma unroll
            for (int j = 0; j < 8; ++j) {
                int row = rbase + j;                 // = b*512 + t
                int b   = row >> 9;
                int t   = row & 511;
                pre[((t << 7) + b) * RNN_H + col] = a[j] + bv;
            }
        }
    }
}

// ---------------------------------------------------------------------------
// Kernel 4: recurrent scan. 8 persistent WGs, each owns 16 batch rows.
// ---------------------------------------------------------------------------
__global__ __launch_bounds__(256)
void rnn_scan_kernel(const float* __restrict__ pre,
                     const unsigned short* __restrict__ Whbf,
                     const long long* __restrict__ lengths,
                     const float* __restrict__ Who,
                     const float* __restrict__ bho,
                     float* __restrict__ out) {
    __shared__ float          hf[16 * HSTRIDE];      // fp32 h (output head)
    __shared__ unsigned short hb[16 * HBSTRIDE];     // bf16 h (WMMA A source)
    __shared__ int            slen[16];

    int tid  = threadIdx.x;
    int lane = tid & 31;
    int w    = tid >> 5;
    int b0   = blockIdx.x * 16;

    for (int i = tid; i < 16 * HSTRIDE;  i += 256) hf[i] = 0.f;
    for (int i = tid; i < 16 * HBSTRIDE; i += 256) hb[i] = 0;
    if (tid < 16) slen[tid] = (int)lengths[b0 + tid];
    __syncthreads();

    int n0 = w * 64;                                 // this wave's feature slice
    int rbase = (lane >> 4) << 3;

    for (int t = 0; t < RNN_T; ++t) {
        v8f acc0 = {}, acc1 = {}, acc2 = {}, acc3 = {};
        #pragma unroll 2
        for (int k0 = 0; k0 < RNN_H; k0 += 32) {
            v16bf aF = load_fragA_lds(hb, k0, lane);
            acc0 = wmma_bf16(aF, load_fragB_bf16(Whbf, n0,      k0, lane), acc0);
            acc1 = wmma_bf16(aF, load_fragB_bf16(Whbf, n0 + 16, k0, lane), acc1);
            acc2 = wmma_bf16(aF, load_fragB_bf16(Whbf, n0 + 32, k0, lane), acc2);
            acc3 = wmma_bf16(aF, load_fragB_bf16(Whbf, n0 + 48, k0, lane), acc3);
        }

        // Prefetch this step's pre values (global, no hazard with hb/hf).
        const float* pret = pre + ((size_t)t * RNN_B + b0) * RNN_H;
        float pv[4][8];
        #pragma unroll
        for (int nt = 0; nt < 4; ++nt) {
            int col = n0 + nt * 16 + (lane & 15);
            #pragma unroll
            for (int j = 0; j < 8; ++j)
                pv[nt][j] = pret[(rbase + j) * RNN_H + col];
        }

        __syncthreads();                             // all reads of h done

        v8f* accs[4] = { &acc0, &acc1, &acc2, &acc3 };
        #pragma unroll
        for (int nt = 0; nt < 4; ++nt) {
            int col = n0 + nt * 16 + (lane & 15);
            v8f a = *accs[nt];
            #pragma unroll
            for (int j = 0; j < 8; ++j) {
                int row = rbase + j;
                if (t < slen[row]) {                 // masked update == keep old h
                    float v = tanhf(a[j] + pv[nt][j]);
                    hf[row * HSTRIDE  + col] = v;
                    hb[row * HBSTRIDE + col] = f2bf(v);
                }
            }
        }
        __syncthreads();                             // h_{t+1} visible
    }

    // Output head: out[b0+row, oc] = h[row,:] . Who[oc,:] + bho[oc]
    if (tid < 128) {
        int row = tid >> 3, oc = tid & 7;
        const float* hr = hf + row * HSTRIDE;
        const float* wr = Who + oc * RNN_H;
        float s = 0.f;
        #pragma unroll 4
        for (int k = 0; k < RNN_H; ++k) s += hr[k] * wr[k];
        out[(b0 + row) * RNN_OUT + oc] = s + bho[oc];
    }
}

// ---------------------------------------------------------------------------
extern "C" void kernel_launch(void* const* d_in, const int* in_sizes, int n_in,
                              void* d_out, int out_size, void* d_ws, size_t ws_size,
                              hipStream_t stream) {
    const float*     x       = (const float*)d_in[0];
    const long long* lengths = (const long long*)d_in[1];
    const float*     We      = (const float*)d_in[2];
    const float*     Wih     = (const float*)d_in[3];
    const float*     bih     = (const float*)d_in[4];
    const float*     Who     = (const float*)d_in[5];
    const float*     bho     = (const float*)d_in[6];
    float*           out     = (float*)d_out;

    char* ws = (char*)d_ws;
    float*          pre  = (float*)ws;                              // 128 MB
    unsigned short* xbf  = (unsigned short*)(ws + 134217728);       //  64 MB
    unsigned short* Wbf  = (unsigned short*)(ws + 201326592);       // 512 KB
    unsigned short* Whbf = (unsigned short*)(ws + 201850880);       // 512 KB

    fuse_weights_kernel<<<(RNN_H * RNN_IN) / 256, 256, 0, stream>>>(Wih, We, Wbf, Whbf);

    const int NX = RNN_B * RNN_T * RNN_IN;                          // 33.5M
    cvt_x_kernel<<<(NX / 8 + 255) / 256, 256, 0, stream>>>(x, xbf, NX / 8);

    gemm_pre_kernel<<<dim3(RNN_H / 64, (RNN_B * RNN_T) / 128), 256, 0, stream>>>(
        xbf, Wbf, bih, pre);

    rnn_scan_kernel<<<RNN_B / 16, 256, 0, stream>>>(pre, Whbf, lengths, Who, bho, out);
}
